// Model_12472585027584
// MI455X (gfx1250) — matrix operand order repaired and checked by reading, not compiled, not run on hardware
//
#include <hip/hip_runtime.h>
#include <hip/hip_bf16.h>
#include <stdint.h>

typedef __attribute__((ext_vector_type(16))) _Float16 v16h;
typedef __attribute__((ext_vector_type(8)))  float    v8f;

#define N_NODES 50000
#define N_EDGES 800000
#define HID 64

// ---- monotone float <-> u32 encoding so unsigned atomicMax == float max ----
__device__ __forceinline__ unsigned enc_f32(float f) {
  unsigned u = __float_as_uint(f);
  return (u & 0x80000000u) ? ~u : (u | 0x80000000u);
}
__device__ __forceinline__ float dec_f32(unsigned u) {
  return (u & 0x80000000u) ? __uint_as_float(u ^ 0x80000000u) : __uint_as_float(~u);
}
// any finite float encodes to >= 0x00800000 (= enc(-FLT_MAX)); init-to-0 means "empty"
#define ENC_EMPTY_LIM 0x00800000u

// ---- fake_quant_w: per-tensor symmetric int8, forward value = q ----
// block 0: W0 [64,4]; blocks 1..4: Ws[b-1] [64,67]. Also emit fp16 Wx (k<64) for WMMA.
__global__ void quant_kernel(const float* __restrict__ W0, const float* __restrict__ Ws,
                             float* __restrict__ Wq0F, float* __restrict__ WqsF,
                             _Float16* __restrict__ WqsH) {
  int b = blockIdx.x, t = threadIdx.x;
  int kdim = (b == 0) ? 4 : 67;
  int nel = 64 * kdim;
  const float* src = (b == 0) ? W0 : Ws + (size_t)(b - 1) * 64 * 67;
  float* dstF      = (b == 0) ? Wq0F : WqsF + (size_t)(b - 1) * 64 * 67;
  __shared__ float red[256];
  float m = 0.f;
  for (int i = t; i < nel; i += 256) m = fmaxf(m, fabsf(src[i]));
  red[t] = m; __syncthreads();
  for (int o = 128; o > 0; o >>= 1) { if (t < o) red[t] = fmaxf(red[t], red[t + o]); __syncthreads(); }
  float s = fmaxf(red[0] / 127.f, 1e-8f);
  for (int i = t; i < nel; i += 256) {
    float q = fminf(fmaxf(rintf(src[i] / s), -127.f), 127.f) * s;  // rne, like jnp.round
    dstF[i] = q;
    if (b > 0) {
      int k = i % kdim;
      if (k < 64) { int c = i / kdim; WqsH[(size_t)(b - 1) * 4096 + c * 64 + k] = (_Float16)q; }
    }
  }
}

__global__ void cvt_f16(const float* __restrict__ in, _Float16* __restrict__ out, int n) {
  int i = blockIdx.x * 256 + threadIdx.x;
  if (i < n) out[i] = (_Float16)in[i];
}

// P[n][c] = pos[n] . Wp[c]  (Wp = columns koff..koff+2 of quantized W, row stride kdim)
__global__ void pos_p_kernel(const float* __restrict__ pos, const float* __restrict__ wq,
                             int kdim, int koff, float* __restrict__ P) {
  int i = blockIdx.x * 256 + threadIdx.x;           // i = n*64 + c
  if (i >= N_NODES * 64) return;
  int n = i >> 6, c = i & 63;
  const float* w = wq + c * kdim + koff;
  P[i] = pos[n * 3] * w[0] + pos[n * 3 + 1] * w[1] + pos[n * 3 + 2] * w[2];
}

// layer 0: a = x (C=1) * Wx + P   (rank-1, no GEMM needed)
__global__ void a0_kernel(const float* __restrict__ x, const float* __restrict__ Wq0F,
                          const float* __restrict__ P, float* __restrict__ aF) {
  int i = blockIdx.x * 256 + threadIdx.x;
  if (i >= N_NODES * 64) return;
  int n = i >> 6, c = i & 63;
  aF[i] = x[n] * Wq0F[c * 4] + P[i];
}

// segment-max: one wave (32 lanes, 2 ch each) per edge; L2-resident atomics.
__global__ void edge_max_kernel(const int* __restrict__ src, const int* __restrict__ dst,
                                const float* __restrict__ aF, unsigned* __restrict__ enc) {
  int t = blockIdx.x * 256 + threadIdx.x;
  int e = t >> 5;
  if (e >= N_EDGES) return;
  int lane = t & 31;
  int j = src[e], i = dst[e];
  const float2* ar = (const float2*)(aF + (size_t)j * 64);
  unsigned* er = enc + (size_t)i * 64 + lane * 2;
  float2 v = ar[lane];
  unsigned e0 = enc_f32(v.x), e1 = enc_f32(v.y);
  if (e0 > er[0]) atomicMax(er,     e0);   // read-filter skips most redundant atomics
  if (e1 > er[1]) atomicMax(er + 1, e1);
}

// h = where(empty,0, max_a - P); LayerNorm(64); ReLU; write z[:,l*64:] (f32) + hH (f16)
__global__ void ln64_kernel(const unsigned* __restrict__ enc, const float* __restrict__ P,
                            const float* __restrict__ ln_g, const float* __restrict__ ln_b,
                            int layer, float* __restrict__ z, _Float16* __restrict__ hH) {
  int node = blockIdx.x * 8 + (threadIdx.x >> 5);
  if (node >= N_NODES) return;
  int lane = threadIdx.x & 31;
  size_t base = (size_t)node * 64 + lane * 2;
  unsigned e0 = enc[base], e1 = enc[base + 1];
  float v0 = (e0 < ENC_EMPTY_LIM) ? 0.f : dec_f32(e0) - P[base];
  float v1 = (e1 < ENC_EMPTY_LIM) ? 0.f : dec_f32(e1) - P[base + 1];
  float s = v0 + v1, ss = v0 * v0 + v1 * v1;
  for (int o = 16; o > 0; o >>= 1) { s += __shfl_xor(s, o); ss += __shfl_xor(ss, o); }
  float mean = s * (1.f / 64.f);
  float var  = ss * (1.f / 64.f) - mean * mean;
  float rs = rsqrtf(var + 1e-5f);
  int c0 = lane * 2;
  float y0 = fmaxf((v0 - mean) * rs * ln_g[layer * 64 + c0]     + ln_b[layer * 64 + c0],     0.f);
  float y1 = fmaxf((v1 - mean) * rs * ln_g[layer * 64 + c0 + 1] + ln_b[layer * 64 + c0 + 1], 0.f);
  z[(size_t)node * 320 + layer * 64 + c0]     = y0;
  z[(size_t)node * 320 + layer * 64 + c0 + 1] = y1;
  hH[base] = (_Float16)y0;
  hH[base + 1] = (_Float16)y1;
}

// LayerNorm(320) with g=1,b=0, emit fp16 for WMMA MLP
__global__ void ln320_kernel(const float* __restrict__ z, _Float16* __restrict__ zH) {
  int node = blockIdx.x * 8 + (threadIdx.x >> 5);
  if (node >= N_NODES) return;
  int lane = threadIdx.x & 31;
  const float* row = z + (size_t)node * 320;
  float v[10], s = 0.f, ss = 0.f;
  for (int t = 0; t < 10; t++) { float x = row[lane + 32 * t]; v[t] = x; s += x; ss += x * x; }
  for (int o = 16; o > 0; o >>= 1) { s += __shfl_xor(s, o); ss += __shfl_xor(ss, o); }
  float mean = s * (1.f / 320.f);
  float rs = rsqrtf(ss * (1.f / 320.f) - mean * mean + 1e-5f);
  _Float16* orow = zH + (size_t)node * 320;
  for (int t = 0; t < 10; t++) orow[lane + 32 * t] = (_Float16)((v[t] - mean) * rs);
}

// Generic WMMA GEMM: out[M,Nout] = act(A[M,K]f16 @ W[Nout,K]f16^T + bias + addF)
// grid = (M/16, Nout/64), block = 128 (4 waves, each owns one 16x16 N-tile).
// Fragment packing per CDNA5 ISA 7.12.2 (16-bit A 16x32, 16-bit B 32x16, f32 C/D).
__global__ void gemm_wmma_kernel(const _Float16* __restrict__ A, const _Float16* __restrict__ W,
                                 const float* __restrict__ bias, const float* __restrict__ addF,
                                 _Float16* __restrict__ outH, float* __restrict__ outF,
                                 int K, int Nout, int relu) {
  int wave = threadIdx.x >> 5, lane = threadIdx.x & 31;
  int m0 = blockIdx.x * 16;
  int n0 = blockIdx.y * 64 + wave * 16;
  int lg = lane >> 4, lm = lane & 15;
  // A: lane(lg,lm) -> row m0+lm, K chunks [8lg,8lg+8) and [8lg+16,8lg+24)
  const _Float16* ap0 = A + (size_t)(m0 + lm) * K + 8 * lg;
  // B: lane(lg,lm) -> column n0+lm == row of W, K chunks [8lg,8lg+8) and [8lg+16,8lg+24)
  const _Float16* bp0 = W + (size_t)(n0 + lm) * K + 8 * lg;
  v8f acc = {};
  union Frag { uint4 u[2]; v16h h; };
  for (int k0 = 0; k0 < K; k0 += 32) {
    Frag a, b;
    a.u[0] = *(const uint4*)(ap0 + k0);
    a.u[1] = *(const uint4*)(ap0 + k0 + 16);
    b.u[0] = *(const uint4*)(bp0 + k0);
    b.u[1] = *(const uint4*)(bp0 + k0 + 16);
    acc = __builtin_amdgcn_wmma_f32_16x16x32_f16(false, a.h, false, b.h,
                                                 (short)0, acc, false, false);
  }
  int colN = n0 + lm;
  float bv = bias ? bias[colN] : 0.f;
  for (int r = 0; r < 8; r++) {
    int row = m0 + r + 8 * lg;          // D layout: VGPR r, lanes 16-31 -> M = r+8
    float v = acc[r] + bv;
    if (addF) v += addF[(size_t)row * Nout + colN];
    if (relu) v = fmaxf(v, 0.f);
    size_t o = (size_t)row * Nout + colN;
    if (outF) outF[o] = v;
    if (outH) outH[o] = (_Float16)v;
  }
}

// out[n] = (t3[n] @ w4^T + b4) * scale   (Nout=2: VALU)
__global__ void final_kernel(const _Float16* __restrict__ t3, const float* __restrict__ w4,
                             const float* __restrict__ b4, const float* __restrict__ scale,
                             float* __restrict__ out) {
  int n = blockIdx.x * 256 + threadIdx.x;
  if (n >= N_NODES) return;
  const _Float16* r = t3 + (size_t)n * 64;
  float a0 = b4[0], a1 = b4[1];
  for (int c = 0; c < 64; c++) {
    float t = (float)r[c];
    a0 += t * w4[c];
    a1 += t * w4[64 + c];
  }
  out[n * 2]     = a0 * scale[0];
  out[n * 2 + 1] = a1 * scale[1];
}

extern "C" void kernel_launch(void* const* d_in, const int* in_sizes, int n_in,
                              void* d_out, int out_size, void* d_ws, size_t ws_size,
                              hipStream_t stream) {
  (void)in_sizes; (void)n_in; (void)out_size; (void)ws_size;
  const float* x    = (const float*)d_in[0];
  const float* pos  = (const float*)d_in[1];
  const int*   ei   = (const int*)d_in[2];
  const float* W0   = (const float*)d_in[3];
  const float* Ws   = (const float*)d_in[4];
  const float* ln_g = (const float*)d_in[5];
  const float* ln_b = (const float*)d_in[6];
  const float* w1 = (const float*)d_in[7];  const float* b1 = (const float*)d_in[8];
  const float* w2 = (const float*)d_in[9];  const float* b2 = (const float*)d_in[10];
  const float* w3 = (const float*)d_in[11]; const float* b3 = (const float*)d_in[12];
  const float* w4 = (const float*)d_in[13]; const float* b4 = (const float*)d_in[14];
  const float* scale = (const float*)d_in[15];
  float* out = (float*)d_out;

  // ---- workspace carve (~142 MB; MLP temporaries alias dead conv buffers) ----
  char* p = (char*)d_ws;
  auto carve = [&](size_t bytes) -> char* {
    char* r = p; p += (bytes + 255) & ~(size_t)255; return r;
  };
  float*    z    = (float*)carve((size_t)N_NODES * 320 * 4);   // 64 MB embeddings concat
  _Float16* zH   = (_Float16*)carve((size_t)N_NODES * 320 * 2);// 32 MB
  float*    aF   = (float*)carve((size_t)N_NODES * 64 * 4);    // 12.8 MB
  float*    P    = (float*)carve((size_t)N_NODES * 64 * 4);    // 12.8 MB
  unsigned* enc  = (unsigned*)carve((size_t)N_NODES * 64 * 4); // 12.8 MB
  _Float16* hH   = (_Float16*)carve((size_t)N_NODES * 64 * 2); // 6.4 MB
  float*    Wq0F = (float*)carve(64 * 4 * 4);
  float*    WqsF = (float*)carve(4 * 64 * 67 * 4);
  _Float16* WqsH = (_Float16*)carve(4 * 64 * 64 * 2);
  _Float16* w1h  = (_Float16*)carve(128 * 320 * 2);
  _Float16* w2h  = (_Float16*)carve(128 * 128 * 2);
  _Float16* w3h  = (_Float16*)carve(64 * 128 * 2);
  _Float16* t1 = (_Float16*)aF;   // alias: conv buffers dead during MLP phase
  _Float16* t2 = (_Float16*)P;
  _Float16* t3 = (_Float16*)enc;

  const int* srcI = ei;
  const int* dstI = ei + N_EDGES;
  const int nc = N_NODES * 64;

  quant_kernel<<<5, 256, 0, stream>>>(W0, Ws, Wq0F, WqsF, WqsH);
  cvt_f16<<<(128 * 320 + 255) / 256, 256, 0, stream>>>(w1, w1h, 128 * 320);
  cvt_f16<<<(128 * 128 + 255) / 256, 256, 0, stream>>>(w2, w2h, 128 * 128);
  cvt_f16<<<(64 * 128 + 255) / 256, 256, 0, stream>>>(w3, w3h, 64 * 128);

  for (int l = 0; l < 5; l++) {
    if (l == 0) {
      pos_p_kernel<<<nc / 256, 256, 0, stream>>>(pos, Wq0F, 4, 1, P);
      a0_kernel<<<nc / 256, 256, 0, stream>>>(x, Wq0F, P, aF);
    } else {
      pos_p_kernel<<<nc / 256, 256, 0, stream>>>(pos, WqsF + (size_t)(l - 1) * 64 * 67, 67, 64, P);
      gemm_wmma_kernel<<<dim3(N_NODES / 16, 1), 128, 0, stream>>>(
          hH, WqsH + (size_t)(l - 1) * 4096, nullptr, P, nullptr, aF, 64, 64, 0);
    }
    hipMemsetAsync(enc, 0, (size_t)nc * 4, stream);   // 0 == "empty segment" sentinel
    edge_max_kernel<<<(N_EDGES * 32) / 256, 256, 0, stream>>>(srcI, dstI, aF, enc);
    ln64_kernel<<<N_NODES / 8, 256, 0, stream>>>(enc, P, ln_g, ln_b, l, z, hH);
  }

  ln320_kernel<<<N_NODES / 8, 256, 0, stream>>>(z, zH);
  gemm_wmma_kernel<<<dim3(N_NODES / 16, 2), 128, 0, stream>>>(zH, w1h, b1, nullptr, t1, nullptr, 320, 128, 1);
  gemm_wmma_kernel<<<dim3(N_NODES / 16, 2), 128, 0, stream>>>(t1, w2h, b2, nullptr, t2, nullptr, 128, 128, 1);
  gemm_wmma_kernel<<<dim3(N_NODES / 16, 1), 128, 0, stream>>>(t2, w3h, b3, nullptr, t3, nullptr, 128, 64, 1);
  final_kernel<<<(N_NODES + 255) / 256, 256, 0, stream>>>(t3, w4, b4, scale, out);
}
